// TextMelAttention_6786048328040
// MI455X (gfx1250) — matrix operand order repaired and checked by reading, not compiled, not run on hardware
//
#include <hip/hip_runtime.h>

// ---------- CDNA5 WMMA types ----------
typedef __attribute__((ext_vector_type(16))) __bf16 v16bf;
typedef __attribute__((ext_vector_type(8)))  float  v8f;

union Frag16 { int i[8]; v16bf v; };

// f32 -> bf16 round-to-nearest-even
__device__ __forceinline__ unsigned short f2bf(float f) {
    unsigned int u = __builtin_bit_cast(unsigned int, f);
    u += 0x7FFFu + ((u >> 16) & 1u);
    return (unsigned short)(u >> 16);
}

// Wave-relative LDS byte offset of a shared-memory pointer.
// On gfx1250 a generic pointer to LDS is {shared_aperture_hi32, lds_offset32},
// so truncation yields the offset the async-DMA ops expect in their LDS-addr VGPR.
__device__ __forceinline__ unsigned lds_off(const void* p) {
    return (unsigned)(size_t)p;
}

// Async global->LDS copy of 16 bytes per active lane (tracked by ASYNCcnt).
__device__ __forceinline__ void async_b128(unsigned lds_addr, const void* gaddr) {
    asm volatile("global_load_async_to_lds_b128 %0, %1, off"
                 :: "v"(lds_addr), "v"((unsigned long long)(size_t)gaddr)
                 : "memory");
}
__device__ __forceinline__ void async_wait0() {
    asm volatile("s_wait_asynccnt 0x0" ::: "memory");
}

// ---------------- prep kernels ----------------
__global__ void cvt_pad_rows(const float* __restrict__ src, unsigned short* __restrict__ dst,
                             int rows, int cin, int cp) {
    long i = (long)blockIdx.x * 256 + threadIdx.x;
    long n = (long)rows * cp;
    if (i >= n) return;
    int c = (int)(i % cp);
    long r = i / cp;
    dst[i] = (c < cin) ? f2bf(src[r * cin + c]) : (unsigned short)0;
}

// src: [cout][cin][taps] f32 -> dst: [cout][tap*cp + ci] bf16 (zero-padded ci>=cin)
__global__ void cvt_weight(const float* __restrict__ src, unsigned short* __restrict__ dst,
                           int cout, int cin, int taps, int cp) {
    long i = (long)blockIdx.x * 256 + threadIdx.x;
    long n = (long)cout * taps * cp;
    if (i >= n) return;
    int per = taps * cp;
    int co  = (int)(i / per);
    int rem = (int)(i % per);
    int tap = rem / cp;
    int ci  = rem % cp;
    dst[i] = (ci < cin) ? f2bf(src[((long)co * cin + ci) * taps + tap]) : (unsigned short)0;
}

// ---------------- generic conv-as-GEMM kernel ----------------
// X:   [B*T][Cp]  bf16 rows; W: [384][Ktot] bf16 (K = tap*Cp + cin); Out: [B*T][384] bf16.
// One wave32 per block, 16 output rows; 12 iterations of dual 16-wide N tiles.
__global__ __launch_bounds__(32)
void conv_gemm_kernel(const unsigned short* __restrict__ X,
                      const unsigned short* __restrict__ W,
                      const float* __restrict__ bias,
                      unsigned short* __restrict__ Out,
                      float* __restrict__ norm,
                      int T, int Cp, int taps, int relu) {
    __shared__ unsigned short sX[18 * 384];
    __shared__ float sNorm[16];

    const int lane = threadIdx.x;
    const int tilesPerBatch = T >> 4;
    const int b  = blockIdx.x / tilesPerBatch;
    const int t0 = (blockIdx.x % tilesPerBatch) << 4;

    const int Ktot   = taps * Cp;
    const int ksteps = Ktot >> 5;          // K chunks of 32
    const int spt    = Cp >> 5;            // K-steps per tap
    const int halo   = (taps == 3) ? 1 : 0;
    const int rowsLd = 16 + 2 * halo;
    const int rowB   = Cp * 2;             // bytes per staged row
    const int cints  = Cp >> 1;

    int* sXi = (int*)sX;
    const unsigned sXbase = lds_off(sX);

    // Stage rows t0-halo .. t0+15+halo: zero-fill out-of-range rows with ds stores,
    // DMA valid rows straight into LDS (no VGPR round trip).
    for (int r = 0; r < rowsLd; ++r) {
        int t = t0 + r - halo;
        if (t < 0 || t >= T) {
            for (int c = lane; c < cints; c += 32) sXi[r * cints + c] = 0;
        } else {
            const char* gsrc = (const char*)(X + ((long)b * T + t) * Cp);
            for (int cb = lane * 16; cb < rowB; cb += 512)
                async_b128(sXbase + (unsigned)(r * rowB + cb), gsrc + cb);
        }
    }
    if (norm && lane < 16) sNorm[lane] = 0.0f;
    async_wait0();
    __syncthreads();

    const int nlo = lane & 15;
    const int h8  = lane >> 4;
    const int* __restrict__ Wi = (const int*)W;

    for (int np = 0; np < 12; ++np) {
        const int n0 = (np << 5) + nlo;
        const int n1 = n0 + 16;
        if (np < 11)
            __builtin_prefetch(W + (size_t)(n0 + 32) * Ktot, 0, 0);

        v8f acc0 = {}, acc1 = {};
        const long wb0 = (((long)n0 * Ktot) >> 1) + h8 * 4;
        const long wb1 = (((long)n1 * Ktot) >> 1) + h8 * 4;
        for (int ks = 0; ks < ksteps; ++ks) {
            const int tap  = ks / spt;
            const int cinb = (ks << 5) - tap * Cp;
            const int abase = (((nlo + tap) * Cp + cinb) >> 1) + h8 * 4;
            Frag16 a, b0, b1;
            #pragma unroll
            for (int i = 0; i < 4; ++i) {
                a.i[i]     = sXi[abase + i];
                a.i[i + 4] = sXi[abase + 8 + i];
            }
            #pragma unroll
            for (int i = 0; i < 8; ++i) b0.i[i] = Wi[wb0 + ks * 16 + i + ((i >> 2) << 2)];
            #pragma unroll
            for (int i = 0; i < 8; ++i) b1.i[i] = Wi[wb1 + ks * 16 + i + ((i >> 2) << 2)];

            acc0 = __builtin_amdgcn_wmma_f32_16x16x32_bf16(
                false, a.v, false, b0.v, (short)0, acc0, false, false);
            acc1 = __builtin_amdgcn_wmma_f32_16x16x32_bf16(
                false, a.v, false, b1.v, (short)0, acc1, false, false);
        }
        // epilogue in f32: bias (+relu), bf16 store, row-norm accumulation
        const float bn0 = bias[n0], bn1 = bias[n1];
        #pragma unroll
        for (int r = 0; r < 8; ++r) {
            const int m = r + (h8 << 3);
            const long orow = ((long)(b * T + t0 + m)) * 384;
            float v0 = acc0[r] + bn0;
            float v1 = acc1[r] + bn1;
            if (relu) { v0 = fmaxf(v0, 0.0f); v1 = fmaxf(v1, 0.0f); }
            Out[orow + n0] = f2bf(v0);
            Out[orow + n1] = f2bf(v1);
            if (norm) atomicAdd(&sNorm[m], v0 * v0 + v1 * v1);   // ds_add_f32
        }
    }
    if (norm) {
        __syncthreads();
        if (lane < 16) norm[b * T + t0 + lane] = sNorm[lane];
    }
}

// ---------------- distance + log-softmax kernel ----------------
// score[b,m,t] = -sqrt(max(||y||^2+||h||^2-2 y.h, 1e-12)), masked, log_softmax over t.
__global__ __launch_bounds__(32)
void dist_softmax_kernel(const unsigned short* __restrict__ Yf,  // [32*1600][384] bf16
                         const unsigned short* __restrict__ Hf,  // [32*400][384] bf16
                         const float* __restrict__ ynorm,        // [32*1600]
                         const float* __restrict__ hnorm,        // [32*400]
                         const unsigned char* __restrict__ mask, // [32*400]
                         float* __restrict__ out) {              // [32][1600][400]
    __shared__ float sS[16 * 400];
    __shared__ unsigned short sY[16 * 384];
    __shared__ float sLse[16];

    const int lane = threadIdx.x;
    const int b  = blockIdx.x / 100;
    const int m0 = (blockIdx.x % 100) << 4;
    const int nlo = lane & 15;
    const int h8  = lane >> 4;
    const int* __restrict__ sYi = (const int*)sY;
    const int* __restrict__ Hi  = (const int*)Hf;

    // stage 16x384 bf16 y-tile once via async DMA (read 25x from LDS afterwards)
    const unsigned sYbase = lds_off(sY);
    for (int r = 0; r < 16; ++r) {
        const char* gsrc = (const char*)(Yf + ((long)(b * 1600 + m0 + r)) * 384);
        for (int cb = lane * 16; cb < 768; cb += 512)
            async_b128(sYbase + (unsigned)(r * 768 + cb), gsrc + cb);
    }
    async_wait0();
    __syncthreads();

    for (int ntile = 0; ntile < 25; ++ntile) {
        const int t = (ntile << 4) + nlo;
        v8f acc = {};
        const long hb = (((long)(b * 400 + t) * 384) >> 1) + h8 * 4;
        for (int ks = 0; ks < 12; ++ks) {
            Frag16 a, bf;
            const int abase = ((nlo * 384 + (ks << 5)) >> 1) + h8 * 4;
            #pragma unroll
            for (int i = 0; i < 4; ++i) {
                a.i[i]     = sYi[abase + i];
                a.i[i + 4] = sYi[abase + 8 + i];
            }
            #pragma unroll
            for (int i = 0; i < 8; ++i)
                bf.i[i] = Hi[hb + ks * 16 + i + ((i >> 2) << 2)];
            acc = __builtin_amdgcn_wmma_f32_16x16x32_bf16(
                false, a.v, false, bf.v, (short)0, acc, false, false);
        }
        const float hn = hnorm[b * 400 + t];
        const bool  mk = mask[b * 400 + t] != 0;
        #pragma unroll
        for (int r = 0; r < 8; ++r) {
            const int mr = r + (h8 << 3);
            float sq = ynorm[b * 1600 + m0 + mr] + hn - 2.0f * acc[r];
            float d  = sqrtf(fmaxf(sq, 1e-12f));
            sS[mr * 400 + t] = mk ? -__builtin_inff() : -d;
        }
    }
    __syncthreads();
    if (lane < 16) {
        float mx = -__builtin_inff();
        for (int c = 0; c < 400; ++c) mx = fmaxf(mx, sS[lane * 400 + c]);
        float s = 0.0f;
        for (int c = 0; c < 400; ++c) s += expf(sS[lane * 400 + c] - mx);
        sLse[lane] = mx + logf(s);
    }
    __syncthreads();
    for (int r = 0; r < 16; ++r) {
        const float l = sLse[r];
        float* o = out + ((long)(b * 1600 + m0 + r)) * 400;
        for (int c = lane; c < 400; c += 32) o[c] = sS[r * 400 + c] - l;
    }
}

// ---------------- launch ----------------
static inline size_t alup(size_t x) { return (x + 255) & ~(size_t)255; }

extern "C" void kernel_launch(void* const* d_in, const int* in_sizes, int n_in,
                              void* d_out, int out_size, void* d_ws, size_t ws_size,
                              hipStream_t stream) {
    const float* hs   = (const float*)d_in[0];   // [32,400,384]
    const float* ys   = (const float*)d_in[1];   // [32,1600,80]
    const unsigned char* mask = (const unsigned char*)d_in[2]; // [32,1,400] bool
    const float* t_w1 = (const float*)d_in[3];
    const float* t_b1 = (const float*)d_in[4];
    const float* t_w2 = (const float*)d_in[5];
    const float* t_b2 = (const float*)d_in[6];
    const float* m_w1 = (const float*)d_in[7];
    const float* m_b1 = (const float*)d_in[8];
    const float* m_w2 = (const float*)d_in[9];
    const float* m_b2 = (const float*)d_in[10];
    const float* m_w3 = (const float*)d_in[11];
    const float* m_b3 = (const float*)d_in[12];
    float* out = (float*)d_out;

    const int B = 32, TT = 400, TM = 1600;
    const long RT = (long)B * TT;   // 12800 text rows
    const long RM = (long)B * TM;   // 51200 mel rows

    char* w = (char*)d_ws;
    size_t off = 0;
    auto take = [&](size_t bytes) { char* p = w + off; off = alup(off + bytes); return p; };
    unsigned short* XH  = (unsigned short*)take(RT * 384 * 2);
    unsigned short* XM  = (unsigned short*)take(RM * 96 * 2);
    unsigned short* WT1 = (unsigned short*)take((size_t)384 * 1152 * 2);
    unsigned short* WT2 = (unsigned short*)take((size_t)384 * 384 * 2);
    unsigned short* WM1 = (unsigned short*)take((size_t)384 * 288 * 2);
    unsigned short* WM2 = (unsigned short*)take((size_t)384 * 1152 * 2);
    unsigned short* WM3 = (unsigned short*)take((size_t)384 * 384 * 2);
    unsigned short* H1  = (unsigned short*)take(RT * 384 * 2);
    unsigned short* HF  = (unsigned short*)take(RT * 384 * 2);
    float*          HN  = (float*)take(RT * 4);
    unsigned short* Y1  = (unsigned short*)take(RM * 384 * 2);
    unsigned short* Y2  = (unsigned short*)take(RM * 384 * 2);
    unsigned short* YF  = (unsigned short*)take(RM * 384 * 2);
    float*          YN  = (float*)take(RM * 4);
    if (off > ws_size) return;

    // --- prep: bf16 conversions / layouts ---
    {
        long n = RT * 384;
        cvt_pad_rows<<<(n + 255) / 256, 256, 0, stream>>>(hs, XH, (int)RT, 384, 384);
    }
    {
        long n = RM * 96;
        cvt_pad_rows<<<(n + 255) / 256, 256, 0, stream>>>(ys, XM, (int)RM, 80, 96);
    }
    cvt_weight<<<(384 * 1152 + 255) / 256, 256, 0, stream>>>(t_w1, WT1, 384, 384, 3, 384);
    cvt_weight<<<(384 * 384  + 255) / 256, 256, 0, stream>>>(t_w2, WT2, 384, 384, 1, 384);
    cvt_weight<<<(384 * 288  + 255) / 256, 256, 0, stream>>>(m_w1, WM1, 384, 80, 3, 96);
    cvt_weight<<<(384 * 1152 + 255) / 256, 256, 0, stream>>>(m_w2, WM2, 384, 384, 3, 384);
    cvt_weight<<<(384 * 384  + 255) / 256, 256, 0, stream>>>(m_w3, WM3, 384, 384, 1, 384);

    // --- text prenet: conv3+relu, conv1 (+norm) ---
    conv_gemm_kernel<<<(int)(RT / 16), 32, 0, stream>>>(XH, WT1, t_b1, H1, nullptr, TT, 384, 3, 1);
    conv_gemm_kernel<<<(int)(RT / 16), 32, 0, stream>>>(H1, WT2, t_b2, HF, HN,      TT, 384, 1, 0);

    // --- mel prenet: conv3+relu, conv3+relu, conv1 (+norm) ---
    conv_gemm_kernel<<<(int)(RM / 16), 32, 0, stream>>>(XM, WM1, m_b1, Y1, nullptr, TM, 96,  3, 1);
    conv_gemm_kernel<<<(int)(RM / 16), 32, 0, stream>>>(Y1, WM2, m_b2, Y2, nullptr, TM, 384, 3, 1);
    conv_gemm_kernel<<<(int)(RM / 16), 32, 0, stream>>>(Y2, WM3, m_b3, YF, YN,      TM, 384, 1, 0);

    // --- distance + log_softmax ---
    dist_softmax_kernel<<<B * (TM / 16), 32, 0, stream>>>(YF, HF, YN, HN, mask, out);
}
